// DANet_15169824489909
// MI455X (gfx1250) — matrix operand order repaired and checked by reading, not compiled, not run on hardware
//
#include <hip/hip_runtime.h>
#include <hip/hip_bf16.h>

typedef _Float16 h16;
typedef __attribute__((ext_vector_type(16))) _Float16 v16h;
typedef __attribute__((ext_vector_type(8)))  float    v8f;

#define BATCH 4
#define CCH   512
#define NPIX  4096      // 64*64
#define HID   64
#define MQKV  640       // 64(q) + 64(k) + 512(v)

// ---------------- WMMA fragment helpers (CDNA5 16x16x32 f16 layouts) ----------------
// A fragment (16m x 32k), source row-major [M,K], leading dim lda:
//   lane = m + 16*h; halfs [0..15] = row m, K = 16h..16h+15
__device__ inline v16h frag_a_rowmajor(const h16* A, int lda) {
  int lane = threadIdx.x & 31;
  int m = lane & 15, h = (lane >> 4) & 1;
  const h16* r0 = A + (size_t)m * lda + h * 16;
  v16h a;
#pragma unroll
  for (int i = 0; i < 16; ++i) a[i] = r0[i];
  return a;
}

// B fragment (32k x 16n) taken from a TRANSPOSED operand BT stored [N, K] row-major:
//   lane = n + 16*h; halfs [j] = B[16h+j][n] = BT[n][16h+j]  -> contiguous 32B per lane
__device__ inline v16h frag_bT(const h16* BT, int ldbt) {
  int lane = threadIdx.x & 31;
  int n = lane & 15, h = (lane >> 4) & 1;
  const h16* p = BT + (size_t)n * ldbt + h * 16;
  v16h b;
#pragma unroll
  for (int j = 0; j < 16; ++j) b[j] = p[j];
  return b;
}

__device__ inline v8f wmma_f16(v16h a, v16h b, v8f c) {
  return __builtin_amdgcn_wmma_f32_16x16x32_f16(false, a, false, b, (short)0, c,
                                                false, false);
}

// ---------------- K0a: x -> feat16 [C,N] + featT16 [N,C], LDS-tiled transpose -------
__global__ __launch_bounds__(256) void k_cvt_x(const float* __restrict__ x,
                                               h16* __restrict__ feat16,
                                               h16* __restrict__ featT16) {
  __shared__ h16 t[32][33];
  int b = blockIdx.z;
  int n0 = blockIdx.x * 32;   // pixel
  int c0 = blockIdx.y * 32;   // channel
  int tx = threadIdx.x & 31;
  int ty0 = (threadIdx.x >> 5) * 4;
  const float* xb = x + (size_t)b * CCH * NPIX;
  h16* fb  = feat16  + (size_t)b * CCH * NPIX;
  h16* ftb = featT16 + (size_t)b * NPIX * CCH;
#pragma unroll
  for (int i = 0; i < 4; ++i) {
    int ty = ty0 + i;
    h16 hv = (h16)xb[(size_t)(c0 + ty) * NPIX + n0 + tx];   // coalesced in n
    fb[(size_t)(c0 + ty) * NPIX + n0 + tx] = hv;            // coalesced in n
    t[ty][tx] = hv;
  }
  __syncthreads();
#pragma unroll
  for (int i = 0; i < 4; ++i) {
    int ty = ty0 + i;
    ftb[(size_t)(n0 + ty) * CCH + c0 + tx] = t[tx][ty];     // coalesced in c
  }
}

// ---------------- K0b: stack {Wq,Wk,Wv} -> w16 [640,512] f16 ----------------
__global__ __launch_bounds__(256) void k_cvt_w(const float* __restrict__ Wq,
                                               const float* __restrict__ Wk,
                                               const float* __restrict__ Wv,
                                               h16* __restrict__ w16) {
  int i = blockIdx.x * 256 + threadIdx.x;   // over 640*512
  int row = i >> 9, col = i & 511;
  float v = (row < 64) ? Wq[row * CCH + col]
          : (row < 128) ? Wk[(row - 64) * CCH + col]
                        : Wv[(row - 128) * CCH + col];
  w16[i] = (h16)v;
}

// ---------------- K1: qkv = Wqkv(640x512) @ feat(512x4096) per batch ----------------
// rows 0..63   -> qT16 [NPIX, HID];  rows 64..127 -> kT16 [NPIX, HID]
// rows 128..   -> v16  [CCH, NPIX]
__global__ __launch_bounds__(256) void k_qkv_gemm(const h16* __restrict__ w16,
                                                  const h16* __restrict__ featT16,
                                                  h16* __restrict__ qT16,
                                                  h16* __restrict__ kT16,
                                                  h16* __restrict__ v16) {
  int b = blockIdx.y;
  int wave = blockIdx.x * 8 + (threadIdx.x >> 5);  // 320 blocks -> 2560 waves
  const int nblocks = NPIX / 64;                   // 64
  int mt = wave / nblocks, nb = wave - mt * nblocks;
  int m0 = mt * 16, n0 = nb * 64;
  const h16* ftb = featT16 + (size_t)b * NPIX * CCH;   // BT: [N=pixel, K=channel]
  const h16* wrow = w16 + (size_t)m0 * CCH;
  const h16* bt0 = ftb + (size_t)(n0 +  0) * CCH;
  const h16* bt1 = ftb + (size_t)(n0 + 16) * CCH;
  const h16* bt2 = ftb + (size_t)(n0 + 32) * CCH;
  const h16* bt3 = ftb + (size_t)(n0 + 48) * CCH;
  v8f acc[4] = {};
  // software pipeline: prefetch next k-step fragments before issuing current WMMAs
  v16h a  = frag_a_rowmajor(wrow, CCH);
  v16h b0 = frag_bT(bt0, CCH), b1 = frag_bT(bt1, CCH);
  v16h b2 = frag_bT(bt2, CCH), b3 = frag_bT(bt3, CCH);
  for (int k0 = 0; k0 < CCH; k0 += 32) {
    int k1 = (k0 + 32 < CCH) ? (k0 + 32) : 0;      // branch-free wrap on last iter
    v16h an  = frag_a_rowmajor(wrow + k1, CCH);
    v16h nb0 = frag_bT(bt0 + k1, CCH), nb1 = frag_bT(bt1 + k1, CCH);
    v16h nb2 = frag_bT(bt2 + k1, CCH), nb3 = frag_bT(bt3 + k1, CCH);
    acc[0] = wmma_f16(a, b0, acc[0]);
    acc[1] = wmma_f16(a, b1, acc[1]);
    acc[2] = wmma_f16(a, b2, acc[2]);
    acc[3] = wmma_f16(a, b3, acc[3]);
    a = an; b0 = nb0; b1 = nb1; b2 = nb2; b3 = nb3;
  }
  int lane = threadIdx.x & 31;
  int ln = lane & 15, h = (lane >> 4) & 1;
  h16* qTb = qT16 + (size_t)b * NPIX * HID;
  h16* kTb = kT16 + (size_t)b * NPIX * HID;
  h16* vb  = v16  + (size_t)b * CCH * NPIX;
#pragma unroll
  for (int t = 0; t < 4; ++t) {
    int col = n0 + t * 16 + ln;
#pragma unroll
    for (int r = 0; r < 8; ++r) {
      int row = m0 + r + 8 * h;
      h16 val = (h16)acc[t][r];
      if (row < 64)        qTb[(size_t)col * HID + row] = val;
      else if (row < 128)  kTb[(size_t)col * HID + (row - 64)] = val;
      else                 vb[(size_t)(row - 128) * NPIX + col] = val;  // coalesced
    }
  }
}

// ---------------- K2: PAM softmax stats (rowmax, rowsum over n=4096) ----------------
__global__ __launch_bounds__(256) void k_pam_stats(const h16* __restrict__ qT16,
                                                   const h16* __restrict__ kT16,
                                                   float* __restrict__ rowmax,
                                                   float* __restrict__ rowsum) {
  int b = blockIdx.y;
  int wave = blockIdx.x * 8 + (threadIdx.x >> 5);  // 32 blocks -> 256 m-tiles
  int m0 = wave * 16;
  const h16* qTb = qT16 + (size_t)b * NPIX * HID;
  const h16* kTb = kT16 + (size_t)b * NPIX * HID;
  v16h a0 = frag_a_rowmajor(qTb + (size_t)m0 * HID, HID);
  v16h a1 = frag_a_rowmajor(qTb + (size_t)m0 * HID + 32, HID);
  float lmax[8];
#pragma unroll
  for (int r = 0; r < 8; ++r) lmax[r] = -3.0e38f;
  {
    v16h b0 = frag_bT(kTb, HID), b1 = frag_bT(kTb + 32, HID);
    for (int nt = 0; nt < NPIX / 16; ++nt) {
      int nt1 = (nt + 1) & (NPIX / 16 - 1);
      const h16* kp = kTb + (size_t)nt1 * 16 * HID;
      v16h nb0 = frag_bT(kp, HID), nb1 = frag_bT(kp + 32, HID);
      v8f d = {};
      d = wmma_f16(a0, b0, d);
      d = wmma_f16(a1, b1, d);
#pragma unroll
      for (int r = 0; r < 8; ++r) lmax[r] = fmaxf(lmax[r], d[r]);
      b0 = nb0; b1 = nb1;
    }
  }
#pragma unroll
  for (int r = 0; r < 8; ++r)
#pragma unroll
    for (int off = 1; off < 16; off <<= 1)
      lmax[r] = fmaxf(lmax[r], __shfl_xor(lmax[r], off, 32));
  float lsum[8] = {};
  {
    v16h b0 = frag_bT(kTb, HID), b1 = frag_bT(kTb + 32, HID);
    for (int nt = 0; nt < NPIX / 16; ++nt) {
      int nt1 = (nt + 1) & (NPIX / 16 - 1);
      const h16* kp = kTb + (size_t)nt1 * 16 * HID;
      v16h nb0 = frag_bT(kp, HID), nb1 = frag_bT(kp + 32, HID);
      v8f d = {};
      d = wmma_f16(a0, b0, d);
      d = wmma_f16(a1, b1, d);
#pragma unroll
      for (int r = 0; r < 8; ++r) lsum[r] += __expf(d[r] - lmax[r]);
      b0 = nb0; b1 = nb1;
    }
  }
#pragma unroll
  for (int r = 0; r < 8; ++r)
#pragma unroll
    for (int off = 1; off < 16; off <<= 1)
      lsum[r] += __shfl_xor(lsum[r], off, 32);
  int lane = threadIdx.x & 31;
  if ((lane & 15) == 0) {
    int h = lane >> 4;
#pragma unroll
    for (int r = 0; r < 8; ++r) {
      rowmax[(size_t)b * NPIX + m0 + r + 8 * h] = lmax[r];
      rowsum[(size_t)b * NPIX + m0 + r + 8 * h] = lsum[r];
    }
  }
}

// ---------------- K3: PAM output, flash-style: pamT[m,c] = sum_n P[m,n] v[c,n] -------
// One block per 16-pixel m-tile. 8 waves each own 64 channels. Each wave computes its
// 32-n chunk of P = exp(logit-max)/sum into LDS; after a barrier all waves consume all
// 8 chunks (256 n per outer step), so logits are computed exactly once per block.
__global__ __launch_bounds__(256) void k_pam_out(const h16* __restrict__ qT16,
                                                 const h16* __restrict__ kT16,
                                                 const h16* __restrict__ v16,
                                                 const float* __restrict__ rowmax,
                                                 const float* __restrict__ rowsum,
                                                 h16* __restrict__ pamT16) {
  __shared__ h16 Plds[16 * 256];
  int b = blockIdx.y;
  int m0 = blockIdx.x * 16;                 // grid.x = 256
  int w = threadIdx.x >> 5;                 // 0..7
  int lane = threadIdx.x & 31;
  int ln = lane & 15, h = (lane >> 4) & 1;
  const h16* qTb = qT16 + (size_t)b * NPIX * HID;
  const h16* kTb = kT16 + (size_t)b * NPIX * HID;
  const h16* vb  = v16  + (size_t)b * CCH * NPIX;  // BT: [N=channel, K=pixel]
  v16h a0 = frag_a_rowmajor(qTb + (size_t)m0 * HID, HID);
  v16h a1 = frag_a_rowmajor(qTb + (size_t)m0 * HID + 32, HID);
  float rmax[8], rinv[8];
#pragma unroll
  for (int r = 0; r < 8; ++r) {
    int m = m0 + r + 8 * h;
    rmax[r] = rowmax[(size_t)b * NPIX + m];
    rinv[r] = 1.0f / rowsum[(size_t)b * NPIX + m];
  }
  int c0 = w * 64;
  const h16* vt0 = vb + (size_t)(c0 +  0) * NPIX;
  const h16* vt1 = vb + (size_t)(c0 + 16) * NPIX;
  const h16* vt2 = vb + (size_t)(c0 + 32) * NPIX;
  const h16* vt3 = vb + (size_t)(c0 + 48) * NPIX;
  v8f acc[4] = {};
  for (int s = 0; s < 16; ++s) {
    int nbase = s * 256 + w * 32;
#pragma unroll
    for (int half = 0; half < 2; ++half) {
      const h16* kp = kTb + (size_t)(nbase + half * 16) * HID;
      v8f d = {};
      d = wmma_f16(a0, frag_bT(kp, HID), d);
      d = wmma_f16(a1, frag_bT(kp + 32, HID), d);
#pragma unroll
      for (int r = 0; r < 8; ++r) {
        float p = __expf(d[r] - rmax[r]) * rinv[r];
        Plds[(r + 8 * h) * 256 + w * 32 + half * 16 + ln] = (h16)p;
      }
    }
    __syncthreads();
    // consume: software pipeline within the barrier-bounded region
    {
      v16h ap  = frag_a_rowmajor(Plds, 256);
      v16h c0f = frag_bT(vt0 + s * 256, NPIX), c1f = frag_bT(vt1 + s * 256, NPIX);
      v16h c2f = frag_bT(vt2 + s * 256, NPIX), c3f = frag_bT(vt3 + s * 256, NPIX);
#pragma unroll
      for (int j = 0; j < 8; ++j) {
        int j1 = (j + 1) & 7;                      // wraps; wrapped prefetch is dead
        int nr1 = s * 256 + j1 * 32;
        v16h apn  = frag_a_rowmajor(Plds + j1 * 32, 256);
        v16h n0f = frag_bT(vt0 + nr1, NPIX), n1f = frag_bT(vt1 + nr1, NPIX);
        v16h n2f = frag_bT(vt2 + nr1, NPIX), n3f = frag_bT(vt3 + nr1, NPIX);
        acc[0] = wmma_f16(ap, c0f, acc[0]);
        acc[1] = wmma_f16(ap, c1f, acc[1]);
        acc[2] = wmma_f16(ap, c2f, acc[2]);
        acc[3] = wmma_f16(ap, c3f, acc[3]);
        ap = apn; c0f = n0f; c1f = n1f; c2f = n2f; c3f = n3f;
      }
    }
    __syncthreads();
  }
  h16* pb = pamT16 + (size_t)b * NPIX * CCH;
#pragma unroll
  for (int t = 0; t < 4; ++t)
#pragma unroll
    for (int r = 0; r < 8; ++r)
      pb[(size_t)(m0 + r + 8 * h) * CCH + c0 + t * 16 + ln] = (h16)acc[t][r];
}

// ---------------- K4: CAM gram logits E[c,d] = sum_n feat[c,n] feat[d,n] -------------
__global__ __launch_bounds__(256) void k_cam_logits(const h16* __restrict__ feat16,
                                                    float* __restrict__ Ef32) {
  int b = blockIdx.y;
  int wave = blockIdx.x * 8 + (threadIdx.x >> 5);  // 32 blocks -> 256 waves
  const int nblocks = CCH / 64;                    // 8
  int mt = wave / nblocks, nb = wave - mt * nblocks;
  int m0 = mt * 16, n0 = nb * 64;
  const h16* fb = feat16 + (size_t)b * CCH * NPIX;
  const h16* arow = fb + (size_t)m0 * NPIX;
  const h16* bt0 = fb + (size_t)(n0 +  0) * NPIX;
  const h16* bt1 = fb + (size_t)(n0 + 16) * NPIX;
  const h16* bt2 = fb + (size_t)(n0 + 32) * NPIX;
  const h16* bt3 = fb + (size_t)(n0 + 48) * NPIX;
  v8f acc[4] = {};
  v16h a  = frag_a_rowmajor(arow, NPIX);
  v16h b0 = frag_bT(bt0, NPIX), b1 = frag_bT(bt1, NPIX);
  v16h b2 = frag_bT(bt2, NPIX), b3 = frag_bT(bt3, NPIX);
  for (int k0 = 0; k0 < NPIX; k0 += 32) {
    int k1 = (k0 + 32 < NPIX) ? (k0 + 32) : 0;
    v16h an  = frag_a_rowmajor(arow + k1, NPIX);
    v16h nb0 = frag_bT(bt0 + k1, NPIX), nb1 = frag_bT(bt1 + k1, NPIX);
    v16h nb2 = frag_bT(bt2 + k1, NPIX), nb3 = frag_bT(bt3 + k1, NPIX);
    acc[0] = wmma_f16(a, b0, acc[0]);
    acc[1] = wmma_f16(a, b1, acc[1]);
    acc[2] = wmma_f16(a, b2, acc[2]);
    acc[3] = wmma_f16(a, b3, acc[3]);
    a = an; b0 = nb0; b1 = nb1; b2 = nb2; b3 = nb3;
  }
  int lane = threadIdx.x & 31;
  int ln = lane & 15, h = (lane >> 4) & 1;
  float* Eb = Ef32 + (size_t)b * CCH * CCH;
#pragma unroll
  for (int t = 0; t < 4; ++t)
#pragma unroll
    for (int r = 0; r < 8; ++r)
      Eb[(size_t)(m0 + r + 8 * h) * CCH + n0 + t * 16 + ln] = acc[t][r];
}

// ---------------- K5: CAM row softmax (over d=512), f32 -> normalized f16 ------------
__global__ __launch_bounds__(256) void k_cam_softmax(const float* __restrict__ Ef32,
                                                     h16* __restrict__ E16) {
  int row = blockIdx.x * 8 + (threadIdx.x >> 5);   // B*CCH = 2048 rows, grid.x=256
  int lane = threadIdx.x & 31;
  const float* src = Ef32 + (size_t)row * CCH;
  float mx = -3.0e38f;
  for (int i = lane; i < CCH; i += 32) mx = fmaxf(mx, src[i]);
#pragma unroll
  for (int off = 1; off < 32; off <<= 1) mx = fmaxf(mx, __shfl_xor(mx, off, 32));
  float sm = 0.0f;
  for (int i = lane; i < CCH; i += 32) sm += __expf(src[i] - mx);
#pragma unroll
  for (int off = 1; off < 32; off <<= 1) sm += __shfl_xor(sm, off, 32);
  float inv = 1.0f / sm;
  h16* dst = E16 + (size_t)row * CCH;
  for (int i = lane; i < CCH; i += 32) dst[i] = (h16)(__expf(src[i] - mx) * inv);
}

// ---------------- K6: CAM out: out[c,n] = gamma_c * sum_d E[c,d] feat[d,n] -----------
__global__ __launch_bounds__(256) void k_cam_out(const h16* __restrict__ E16,
                                                 const h16* __restrict__ featT16,
                                                 const float* __restrict__ gamma_c,
                                                 float* __restrict__ out) {
  int b = blockIdx.y;
  int wave = blockIdx.x * 8 + (threadIdx.x >> 5);  // 256 blocks -> 2048 waves
  const int nblocks = NPIX / 64;                   // 64
  int mt = wave / nblocks, nb = wave - mt * nblocks;
  int m0 = mt * 16, n0 = nb * 64;
  const h16* Eb  = E16     + (size_t)b * CCH * CCH;
  const h16* ftb = featT16 + (size_t)b * NPIX * CCH;
  const h16* arow = Eb + (size_t)m0 * CCH;
  const h16* bt0 = ftb + (size_t)(n0 +  0) * CCH;
  const h16* bt1 = ftb + (size_t)(n0 + 16) * CCH;
  const h16* bt2 = ftb + (size_t)(n0 + 32) * CCH;
  const h16* bt3 = ftb + (size_t)(n0 + 48) * CCH;
  v8f acc[4] = {};
  v16h a  = frag_a_rowmajor(arow, CCH);
  v16h b0 = frag_bT(bt0, CCH), b1 = frag_bT(bt1, CCH);
  v16h b2 = frag_bT(bt2, CCH), b3 = frag_bT(bt3, CCH);
  for (int k0 = 0; k0 < CCH; k0 += 32) {
    int k1 = (k0 + 32 < CCH) ? (k0 + 32) : 0;
    v16h an  = frag_a_rowmajor(arow + k1, CCH);
    v16h nb0 = frag_bT(bt0 + k1, CCH), nb1 = frag_bT(bt1 + k1, CCH);
    v16h nb2 = frag_bT(bt2 + k1, CCH), nb3 = frag_bT(bt3 + k1, CCH);
    acc[0] = wmma_f16(a, b0, acc[0]);
    acc[1] = wmma_f16(a, b1, acc[1]);
    acc[2] = wmma_f16(a, b2, acc[2]);
    acc[3] = wmma_f16(a, b3, acc[3]);
    a = an; b0 = nb0; b1 = nb1; b2 = nb2; b3 = nb3;
  }
  float g = gamma_c[0];
  int lane = threadIdx.x & 31;
  int ln = lane & 15, h = (lane >> 4) & 1;
  float* ob = out + (size_t)b * CCH * NPIX;
#pragma unroll
  for (int t = 0; t < 4; ++t)
#pragma unroll
    for (int r = 0; r < 8; ++r)
      ob[(size_t)(m0 + r + 8 * h) * NPIX + n0 + t * 16 + ln] = g * acc[t][r];
}

// ---------------- K7: combine: out[c,m] += gamma_p * pamT[m,c] + 2*x[c,m] ------------
__global__ __launch_bounds__(256) void k_combine(const h16* __restrict__ pamT16,
                                                 const float* __restrict__ x,
                                                 const float* __restrict__ gamma_p,
                                                 float* __restrict__ out) {
  __shared__ float tile[16][17];
  int b = blockIdx.z;
  int m0 = blockIdx.x * 16;   // pixel
  int c0 = blockIdx.y * 16;   // channel
  int tx = threadIdx.x & 15, ty = threadIdx.x >> 4;
  const h16* pb = pamT16 + (size_t)b * NPIX * CCH;
  tile[ty][tx] = (float)pb[(size_t)(m0 + ty) * CCH + c0 + tx];   // coalesced in c
  __syncthreads();
  float g = gamma_p[0];
  int c = c0 + ty, m = m0 + tx;
  size_t idx = (size_t)b * CCH * NPIX + (size_t)c * NPIX + m;    // coalesced in m
  out[idx] += g * tile[tx][ty] + 2.0f * x[idx];
}

// ---------------------------------- launcher ----------------------------------------
extern "C" void kernel_launch(void* const* d_in, const int* in_sizes, int n_in,
                              void* d_out, int out_size, void* d_ws, size_t ws_size,
                              hipStream_t stream) {
  const float* x       = (const float*)d_in[0];
  const float* Wq      = (const float*)d_in[1];
  const float* Wk      = (const float*)d_in[2];
  const float* Wv      = (const float*)d_in[3];
  const float* gamma_p = (const float*)d_in[4];
  const float* gamma_c = (const float*)d_in[5];
  float* out = (float*)d_out;

  char* ws = (char*)d_ws;
  size_t off = 0;
  auto carve = [&](size_t bytes) {
    char* p = ws + off;
    off = (off + bytes + 255) & ~(size_t)255;
    return p;
  };
  h16*   feat16  = (h16*)  carve((size_t)BATCH * CCH * NPIX * 2);
  h16*   featT16 = (h16*)  carve((size_t)BATCH * NPIX * CCH * 2);
  h16*   kT16    = (h16*)  carve((size_t)BATCH * NPIX * HID * 2);
  h16*   qT16    = (h16*)  carve((size_t)BATCH * NPIX * HID * 2);
  h16*   v16     = (h16*)  carve((size_t)BATCH * CCH * NPIX * 2);
  h16*   w16     = (h16*)  carve((size_t)MQKV * CCH * 2);
  float* rowmax  = (float*)carve((size_t)BATCH * NPIX * 4);
  float* rowsum  = (float*)carve((size_t)BATCH * NPIX * 4);
  float* Ef32    = (float*)carve((size_t)BATCH * CCH * CCH * 4);
  h16*   E16     = (h16*)  carve((size_t)BATCH * CCH * CCH * 2);
  h16*   pamT16  = (h16*)  carve((size_t)BATCH * NPIX * CCH * 2);
  (void)ws_size; (void)in_sizes; (void)n_in; (void)out_size;

  // 0) precision staging (tiled transpose through LDS)
  k_cvt_x<<<dim3(NPIX / 32, CCH / 32, BATCH), 256, 0, stream>>>(x, feat16, featT16);
  k_cvt_w<<<dim3((MQKV * CCH) / 256), 256, 0, stream>>>(Wq, Wk, Wv, w16);
  // 1) fused QKV projection GEMM (writes attention-ready layouts qT / kT / v)
  k_qkv_gemm<<<dim3(320, BATCH), 256, 0, stream>>>(w16, featT16, qT16, kT16, v16);
  // 2) PAM softmax statistics
  k_pam_stats<<<dim3(32, BATCH), 256, 0, stream>>>(qT16, kT16, rowmax, rowsum);
  // 3) PAM output (flash-style, P staged through LDS, shared by 8 waves)
  k_pam_out<<<dim3(256, BATCH), 256, 0, stream>>>(qT16, kT16, v16, rowmax, rowsum, pamT16);
  // 4) CAM gram matrix
  k_cam_logits<<<dim3(32, BATCH), 256, 0, stream>>>(feat16, Ef32);
  // 5) CAM softmax
  k_cam_softmax<<<dim3(256), 256, 0, stream>>>(Ef32, E16);
  // 6) CAM output (initializes d_out with gamma_c * cam)
  k_cam_out<<<dim3(256, BATCH), 256, 0, stream>>>(E16, featT16, gamma_c, out);
  // 7) combine: += gamma_p * pam (LDS transpose) + 2*x
  k_combine<<<dim3(NPIX / 16, CCH / 16, BATCH), 256, 0, stream>>>(pamT16, x, gamma_p, out);
}
